// LoGNet_60516089201069
// MI455X (gfx1250) — matrix operand order repaired and checked by reading, not compiled, not run on hardware
//
#include <hip/hip_runtime.h>
#include <math.h>

typedef __attribute__((ext_vector_type(16))) _Float16 v16h;
typedef __attribute__((ext_vector_type(8)))  float    v8f;

#define WMMA16(a,b,c) __builtin_amdgcn_wmma_f32_16x16x32_f16(false,(a),false,(b),(short)0,(c),false,false)

// ---------------- WMMA helpers (wave32 layouts per CDNA5 ISA 7.12.2) ----------------

// Build a 16x32 f16 A tile from an LDS f32 [16][ldr] region, K window = kc*32..kc*32+31.
// Lanes 0-15: row=lane, v0..v3 hold K 0..7 (pairs), v4..v7 hold K 16..23.
// Lanes 16-31: row=lane-16, K 8..15 and 24..31.
__device__ __forceinline__ v16h load_A_lds(const float* row0, int ldr, int kc) {
  const int lane = threadIdx.x & 31;
  const int row  = lane & 15;
  const int hi   = (lane >> 4) << 3;          // 0 or 8
  const float* rp = row0 + row * ldr + kc * 32;
  v16h a;
#pragma unroll
  for (int j = 0; j < 8; ++j) {
    int kb = (j < 4) ? (hi + 2 * j) : (16 + hi + 2 * (j - 4));
    a[2 * j]     = (_Float16)rp[kb];
    a[2 * j + 1] = (_Float16)rp[kb + 1];
  }
  return a;
}

// B tiles are pre-packed: per tile 32 lanes x 16 f16, lane-contiguous.
__device__ __forceinline__ v16h load_B_tile(const _Float16* tile) {
  const int lane = threadIdx.x & 31;
  return *(const v16h*)(tile + lane * 16);
}

// C/D f32 layout: VGPR r, lanes 0-15 -> M=r, N=lane; lanes 16-31 -> M=r+8, N=lane-16.
__device__ __forceinline__ void store_C_lds(float* out, int ldc, int ct, v8f c) {
  const int lane = threadIdx.x & 31;
  const int col  = ct * 16 + (lane & 15);
  const int rb   = (lane >> 4) * 8;
#pragma unroll
  for (int r = 0; r < 8; ++r) out[(rb + r) * ldc + col] = c[r];
}

__device__ __forceinline__ void store_C_bias(float* out, int ldc, int ct, v8f c, const float* bias) {
  const int lane = threadIdx.x & 31;
  const int col  = ct * 16 + (lane & 15);
  const int rb   = (lane >> 4) * 8;
  const float b  = bias[col];
#pragma unroll
  for (int r = 0; r < 8; ++r) out[(rb + r) * ldc + col] = c[r] + b;
}

// [16,64] (LDS f32) x [64,160] (packed f16, 20 tiles kc*10+ct) -> LDS [16][160] f32
__device__ __forceinline__ void gemm_gates(const float* Alds, const _Float16* Bt, float* Clds) {
  v16h a0 = load_A_lds(Alds, 64, 0);
  v16h a1 = load_A_lds(Alds, 64, 1);
#pragma unroll
  for (int ct = 0; ct < 10; ++ct) {
    v8f c = {};
    c = WMMA16(a0, load_B_tile(Bt + (0 * 10 + ct) * 512), c);
    c = WMMA16(a1, load_B_tile(Bt + (1 * 10 + ct) * 512), c);
    store_C_lds(Clds, 160, ct, c);
  }
}

__device__ __forceinline__ float sigm(float x) { return 1.f / (1.f + __expf(-x)); }

// ---------------- weight prep kernels ----------------

// Pack W [150,50] f32 (gi[n]=sum_k x[k]*W[n][k]) into B-layout f16, K padded to 64, Ncols to 160.
__global__ void prep_gru_w(const float* __restrict__ w, _Float16* __restrict__ out) {
  int gid = blockIdx.x * blockDim.x + threadIdx.x;
  if (gid >= 20 * 512) return;
  int tile = gid >> 9, rem = gid & 511;
  int lane = rem >> 4, i = rem & 15;
  int kc = tile / 10, ct = tile % 10;
  int n = ct * 16 + (lane & 15);
  int k = kc * 32 + ((i < 8) ? ((lane >> 4) * 8 + i) : (16 + (lane >> 4) * 8 + (i - 8)));
  float v = (k < 50 && n < 150) ? w[n * 50 + k] : 0.f;
  out[gid] = (_Float16)v;
}

// Wc = W_gat @ W_scale  [50,300]
__global__ void prep_wc(const float* __restrict__ wgat, const float* __restrict__ wscale,
                        float* __restrict__ wc) {
  int gid = blockIdx.x * blockDim.x + threadIdx.x;
  if (gid >= 50 * 300) return;
  int r = gid / 300, c = gid - r * 300;
  float s = 0.f;
  for (int j = 0; j < 50; ++j) s += wgat[r * 50 + j] * wscale[j * 300 + c];
  wc[gid] = s;
}

// bc = W_gat @ b_scale (padded to 64)
__global__ void prep_bc(const float* __restrict__ wgat, const float* __restrict__ bscale,
                        float* __restrict__ bc) {
  int r = threadIdx.x;
  if (r >= 64) return;
  float s = 0.f;
  if (r < 50) for (int j = 0; j < 50; ++j) s += wgat[r * 50 + j] * bscale[j];
  bc[r] = s;
}

// Pack Wc into B-layout f16: K padded 300->320 (10 kc), N 50->64 (4 ct), tile = kc*4+ct
__global__ void prep_wc_b(const float* __restrict__ wc, _Float16* __restrict__ out) {
  int gid = blockIdx.x * blockDim.x + threadIdx.x;
  if (gid >= 40 * 512) return;
  int tile = gid >> 9, rem = gid & 511;
  int lane = rem >> 4, i = rem & 15;
  int kc = tile >> 2, ct = tile & 3;
  int n = ct * 16 + (lane & 15);
  int k = kc * 32 + ((i < 8) ? ((lane >> 4) * 8 + i) : (16 + (lane >> 4) * 8 + (i - 8)));
  float v = (k < 300 && n < 50) ? wc[n * 300 + k] : 0.f;
  out[gid] = (_Float16)v;
}

// ---------------- fused per-node pipeline: one wave32 per 16 nodes ----------------
__global__ __launch_bounds__(32)
void fused_node_kernel(const int* __restrict__ kg, const float* __restrict__ emb,
                       const float* __restrict__ bihf, const float* __restrict__ bhhf,
                       const float* __restrict__ bihb, const float* __restrict__ bhhb,
                       const float* __restrict__ lng, const float* __restrict__ lnb,
                       const float* __restrict__ attl, const float* __restrict__ attr,
                       const _Float16* __restrict__ wihf_t, const _Float16* __restrict__ whhf_t,
                       const _Float16* __restrict__ wihb_t, const _Float16* __restrict__ whhb_t,
                       const _Float16* __restrict__ wc_t, const float* __restrict__ bc,
                       float* __restrict__ featw, float* __restrict__ localw,
                       float* __restrict__ elw, float* __restrict__ erw, int N) {
  __shared__ float sm[14400];
  float* s_x   = sm;                 // [3][16][64]
  float* s_h   = sm + 3072;          // [6][16][64]  slots: hf1,hf2,hf3,g1,g2,g3
  float* s_gi  = sm + 3072 + 6144;   // [16][160]
  float* s_gh  = s_gi + 2560;        // [16][160]
  float* s_cat = s_gi;               // [16][320] (reuses gi+gh)
  float* s_feat = s_x;               // [16][64]  (reuses x)
  float* s_mu  = sm + 14336;         // [16]
  float* s_rs  = sm + 14352;         // [16]
  const int lane = threadIdx.x;
  const int base = blockIdx.x * 16;

  for (int idx = lane; idx < 6144; idx += 32) s_h[idx] = 0.f;
  // embedding gather, zero-padded to 64 cols
  for (int idx = lane; idx < 3072; idx += 32) {
    int t = idx >> 10, r = idx & 1023, m = r >> 6, d = r & 63;
    int node = base + m;
    float v = 0.f;
    if (d < 50 && node < N) v = emb[kg[node * 3 + t] * 50 + d];
    s_x[idx] = v;
  }
  __syncthreads();

  // BiGRU: dir 0 = forward (t = step), dir 1 = backward (t = 2-step)
  for (int dir = 0; dir < 2; ++dir) {
    const _Float16* wih = dir ? wihb_t : wihf_t;
    const _Float16* whh = dir ? whhb_t : whhf_t;
    const float* bih = dir ? bihb : bihf;
    const float* bhh = dir ? bhhb : bhhf;
    const int hslot = dir ? 3 : 0;
    for (int step = 0; step < 3; ++step) {
      const int t = dir ? (2 - step) : step;
      const float* xin   = s_x + t * 1024;
      float*       hout  = s_h + (hslot + step) * 1024;
      const float* hprev = (step == 0) ? nullptr : s_h + (hslot + step - 1) * 1024;
      gemm_gates(xin, wih, s_gi);
      if (hprev) gemm_gates(hprev, whh, s_gh);
      __syncthreads();
      for (int idx = lane; idx < 800; idx += 32) {
        int m = idx / 50, d = idx - m * 50;
        const float* gi = s_gi + m * 160;
        float ir = gi[d] + bih[d];
        float iz = gi[50 + d] + bih[50 + d];
        float in_ = gi[100 + d] + bih[100 + d];
        float hr = bhh[d], hz = bhh[50 + d], hn = bhh[100 + d], hp = 0.f;
        if (hprev) {
          const float* gh = s_gh + m * 160;
          hr += gh[d]; hz += gh[50 + d]; hn += gh[100 + d];
          hp = hprev[m * 64 + d];
        }
        float r = sigm(ir + hr);
        float z = sigm(iz + hz);
        float nn = tanhf(in_ + r * hn);
        hout[m * 64 + d] = (1.f - z) * nn + z * hp;
      }
      __syncthreads();
    }
  }

  // local score + LayerNorm stats (lane m handles node m)
  if (lane < 16) {
    const float* h0 = s_h + 0 * 1024 + lane * 64;   // hf1
    const float* h1 = s_h + 1 * 1024 + lane * 64;   // hf2
    const float* h2 = s_h + 2 * 1024 + lane * 64;   // hf3
    const float* g1 = s_h + 3 * 1024 + lane * 64;
    const float* g2 = s_h + 4 * 1024 + lane * 64;
    const float* g3 = s_h + 5 * 1024 + lane * 64;
    float nsum = 0.f, sum = 0.f, sq = 0.f;
    for (int d = 0; d < 50; ++d) {
      float a0 = h0[d], a1 = h1[d], a2 = h2[d];
      float b1 = g1[d], b2 = g2[d], b3 = g3[d];
      float u = a0 + a1 - a2;       // fwd half of s+p-o
      float w = b3 + b2 - b1;       // bwd half (s uses g3, p g2, o g1)
      nsum += u * u + w * w;
      sum += a0 + a1 + a2 + b1 + b2 + b3;
      sq  += a0 * a0 + a1 * a1 + a2 * a2 + b1 * b1 + b2 * b2 + b3 * b3;
    }
    int node = base + lane;
    if (node < N) localw[node] = sigm(sqrtf(nsum));
    float mu = sum * (1.f / 300.f);
    float var = sq * (1.f / 300.f) - mu * mu;
    s_mu[lane] = mu;
    s_rs[lane] = rsqrtf(var + 1e-5f);
  }
  __syncthreads();

  // normalized cat [16][320]: order s=[hf1|g3], p=[hf2|g2], o=[hf3|g1]
  for (int idx = lane; idx < 5120; idx += 32) {
    int m = idx / 320, j = idx - m * 320;
    float v = 0.f;
    if (j < 300) {
      int seg = j / 50, d = j - seg * 50;
      int slot = (seg & 1) ? (5 - (seg >> 1)) : (seg >> 1);  // 0,5,1,4,2,3
      float raw = s_h[slot * 1024 + m * 64 + d];
      v = (raw - s_mu[m]) * s_rs[m] * lng[j] + lnb[j];
    }
    s_cat[idx] = v;
  }
  __syncthreads();

  // feat = LN(cat) @ (W_gat@W_scale).T + W_gat@b_scale   [16,320]x[320,64]
  v8f acc0 = {}, acc1 = {}, acc2 = {}, acc3 = {};
#pragma unroll
  for (int kc = 0; kc < 10; ++kc) {
    v16h a = load_A_lds(s_cat, 320, kc);
    acc0 = WMMA16(a, load_B_tile(wc_t + (kc * 4 + 0) * 512), acc0);
    acc1 = WMMA16(a, load_B_tile(wc_t + (kc * 4 + 1) * 512), acc1);
    acc2 = WMMA16(a, load_B_tile(wc_t + (kc * 4 + 2) * 512), acc2);
    acc3 = WMMA16(a, load_B_tile(wc_t + (kc * 4 + 3) * 512), acc3);
  }
  __syncthreads();                   // done reading s_x alias before feat overwrite
  store_C_bias(s_feat, 64, 0, acc0, bc);
  store_C_bias(s_feat, 64, 1, acc1, bc);
  store_C_bias(s_feat, 64, 2, acc2, bc);
  store_C_bias(s_feat, 64, 3, acc3, bc);
  __syncthreads();

  // attention projections: lanes 0-15 -> el, lanes 16-31 -> er
  {
    int m = lane & 15;
    const float* f = s_feat + m * 64;
    const float* av = (lane < 16) ? attl : attr;
    float s = 0.f;
    for (int d = 0; d < 50; ++d) s += f[d] * av[d];
    int node = base + m;
    if (node < N) { if (lane < 16) elw[node] = s; else erw[node] = s; }
  }
  // feat writeout (padded cols are exact zeros by construction)
  for (int idx = lane; idx < 1024; idx += 32) {
    int m = idx >> 6, d = idx & 63;
    int node = base + m;
    if (node < N) featw[node * 64 + d] = s_feat[idx];
  }
}

// ---------------- edge phase ----------------

__device__ __forceinline__ unsigned fkey(float f) {
  unsigned u = __float_as_uint(f);
  return (u & 0x80000000u) ? ~u : (u | 0x80000000u);
}
__device__ __forceinline__ float funkey(unsigned k) {
  unsigned u = (k & 0x80000000u) ? (k ^ 0x80000000u) : ~k;
  return __uint_as_float(u);
}

__global__ void edge_max(const int* __restrict__ src, const int* __restrict__ dst,
                         const float* __restrict__ el, const float* __restrict__ er,
                         float* __restrict__ ev, unsigned* __restrict__ emaxk, int E) {
  int g = blockIdx.x * blockDim.x + threadIdx.x;
  if (g >= E) return;
  int d = dst[g];
  float v = el[src[g]] + er[d];
  v = (v > 0.f) ? v : 0.2f * v;
  ev[g] = v;
  atomicMax(&emaxk[d], fkey(v));
}

__global__ void edge_exp(const int* __restrict__ dst, const unsigned* __restrict__ emaxk,
                         float* __restrict__ ev, float* __restrict__ denom, int E) {
  int g = blockIdx.x * blockDim.x + threadIdx.x;
  if (g >= E) return;
  float m = funkey(emaxk[dst[g]]);
  if (!isfinite(m)) m = 0.f;
  float e = __expf(ev[g] - m);
  ev[g] = e;
  atomicAdd(&denom[dst[g]], e);
}

__global__ void edge_alpha(const int* __restrict__ dst, const float* __restrict__ denom,
                           float* __restrict__ ev, int E) {
  int g = blockIdx.x * blockDim.x + threadIdx.x;
  if (g >= E) return;
  ev[g] = ev[g] / (denom[dst[g]] + 1e-9f);
}

__global__ void edge_scatter(const int* __restrict__ src, const int* __restrict__ dst,
                             const float* __restrict__ alpha, const float* __restrict__ feat,
                             float* __restrict__ zacc, int E) {
  int g = blockIdx.x * blockDim.x + threadIdx.x;
  int e = g >> 6, d = g & 63;
  if (e >= E || d >= 50) return;
  atomicAdd(&zacc[(long long)dst[e] * 50 + d], alpha[e] * feat[(long long)src[e] * 64 + d]);
}

__global__ void z_final(float* __restrict__ zacc, const float* __restrict__ gbias,
                        const float* __restrict__ localv, float* __restrict__ score, int N) {
  int n = blockIdx.x * blockDim.x + threadIdx.x;
  if (n >= N) return;
  float ss = 0.f;
  float* zp = zacc + (long long)n * 50;
  for (int d = 0; d < 50; ++d) {
    float v = zp[d] + gbias[d];
    zp[d] = v;
    ss += v * v;
  }
  float gt = 1.f / (1.f + __expf(-sqrtf(ss)));
  score[n] = gt - 0.7f * localv[n];
}

__global__ void score_mask(const int* __restrict__ mask, const float* __restrict__ sfull,
                           float* __restrict__ sout, int N) {
  int i = blockIdx.x * blockDim.x + threadIdx.x;
  if (i < N) sout[i] = sfull[mask[i]];
}

__global__ void margin_loss(const int* __restrict__ pidx, const int* __restrict__ nidx,
                            const float* __restrict__ sout, float* __restrict__ loss, int Nh) {
  int i = blockIdx.x * blockDim.x + threadIdx.x;
  if (i >= Nh) return;
  float diff = sout[nidx[i]] - sout[pidx[i]] + 1.f;
  float r = diff > 0.f ? diff : 0.f;
  atomicAdd(loss, r * (1.f / (float)Nh));
}

// ---------------- host side ----------------

extern "C" void kernel_launch(void* const* d_in, const int* in_sizes, int n_in,
                              void* d_out, int out_size, void* d_ws, size_t ws_size,
                              hipStream_t stream) {
  const int*   kg     = (const int*)d_in[0];
  const int*   src    = (const int*)d_in[1];
  const int*   dst    = (const int*)d_in[2];
  const int*   mask   = (const int*)d_in[3];
  const int*   pidx   = (const int*)d_in[4];
  const int*   nidx   = (const int*)d_in[5];
  const float* emb    = (const float*)d_in[6];
  const float* wihf   = (const float*)d_in[7];
  const float* whhf   = (const float*)d_in[8];
  const float* bihf   = (const float*)d_in[9];
  const float* bhhf   = (const float*)d_in[10];
  const float* wihb   = (const float*)d_in[11];
  const float* whhb   = (const float*)d_in[12];
  const float* bihb   = (const float*)d_in[13];
  const float* bhhb   = (const float*)d_in[14];
  const float* lng    = (const float*)d_in[15];
  const float* lnb    = (const float*)d_in[16];
  const float* wscale = (const float*)d_in[17];
  const float* bscale = (const float*)d_in[18];
  const float* wgat   = (const float*)d_in[19];
  const float* attl   = (const float*)d_in[20];
  const float* attr   = (const float*)d_in[21];
  const float* gbias  = (const float*)d_in[22];
  const int N  = in_sizes[3];
  const int E  = in_sizes[1];
  const int Nh = in_sizes[4];

  char* ws = (char*)d_ws;
  size_t off = 0;
  auto alloc = [&](size_t bytes) -> char* {
    char* p = ws + off;
    off = (off + bytes + 255) & ~(size_t)255;
    return p;
  };
  _Float16* wihf_t = (_Float16*)alloc(20 * 512 * 2);
  _Float16* whhf_t = (_Float16*)alloc(20 * 512 * 2);
  _Float16* wihb_t = (_Float16*)alloc(20 * 512 * 2);
  _Float16* whhb_t = (_Float16*)alloc(20 * 512 * 2);
  _Float16* wc_t   = (_Float16*)alloc(40 * 512 * 2);
  float*    wc_raw = (float*)alloc(50 * 300 * 4);
  float*    bc     = (float*)alloc(64 * 4);
  float*    featw  = (float*)alloc((size_t)N * 64 * 4);
  float*    localw = (float*)alloc((size_t)N * 4);
  float*    elw    = (float*)alloc((size_t)N * 4);
  float*    erw    = (float*)alloc((size_t)N * 4);
  float*    ev     = (float*)alloc((size_t)E * 4);
  unsigned* emaxk  = (unsigned*)alloc((size_t)N * 4);
  float*    denom  = (float*)alloc((size_t)N * 4);
  float*    sfull  = (float*)alloc((size_t)N * 4);

  float* out   = (float*)d_out;
  float* loss  = out;
  float* zacc  = out + 1;
  float* sout  = out + 1 + (size_t)N * 50;

  hipMemsetAsync(d_out, 0, (size_t)out_size * sizeof(float), stream);
  hipMemsetAsync(emaxk, 0, (size_t)N * 4, stream);   // key 0 == identity for sortable-uint max
  hipMemsetAsync(denom, 0, (size_t)N * 4, stream);

  // weight prep
  prep_gru_w<<<40, 256, 0, stream>>>(wihf, wihf_t);
  prep_gru_w<<<40, 256, 0, stream>>>(whhf, whhf_t);
  prep_gru_w<<<40, 256, 0, stream>>>(wihb, wihb_t);
  prep_gru_w<<<40, 256, 0, stream>>>(whhb, whhb_t);
  prep_wc<<<(50 * 300 + 255) / 256, 256, 0, stream>>>(wgat, wscale, wc_raw);
  prep_bc<<<1, 64, 0, stream>>>(wgat, bscale, bc);
  prep_wc_b<<<80, 256, 0, stream>>>(wc_raw, wc_t);

  // fused per-node pipeline (WMMA)
  fused_node_kernel<<<(N + 15) / 16, 32, 0, stream>>>(
      kg, emb, bihf, bhhf, bihb, bhhb, lng, lnb, attl, attr,
      wihf_t, whhf_t, wihb_t, whhb_t, wc_t, bc,
      featw, localw, elw, erw, N);

  // GAT edge softmax + aggregation
  int eb = (E + 255) / 256;
  edge_max<<<eb, 256, 0, stream>>>(src, dst, elw, erw, ev, emaxk, E);
  edge_exp<<<eb, 256, 0, stream>>>(dst, emaxk, ev, denom, E);
  edge_alpha<<<eb, 256, 0, stream>>>(dst, denom, ev, E);
  long long sc_threads = (long long)E * 64;
  edge_scatter<<<(unsigned)((sc_threads + 255) / 256), 256, 0, stream>>>(src, dst, ev, featw, zacc, E);

  // finalize
  z_final<<<(N + 255) / 256, 256, 0, stream>>>(zacc, gbias, localw, sfull, N);
  score_mask<<<(N + 255) / 256, 256, 0, stream>>>(mask, sfull, sout, N);
  margin_loss<<<(Nh + 255) / 256, 256, 0, stream>>>(pidx, nidx, sout, loss, Nh);
}
